// GPT2Model_87342454931564
// MI455X (gfx1250) — matrix operand order repaired and checked by reading, not compiled, not run on hardware
//
#include <hip/hip_runtime.h>
#include <hip/hip_bf16.h>
#include <math.h>

// ---------------------------------------------------------------------------
// GPT-2 forward for MI455X (gfx1250): bf16 WMMA GEMMs + flash attention,
// async global->LDS double buffering (ASYNCcnt), WMMA-based softmax row sums.
// L=6, E=768, H=12, D=64, FF=3072, B=2, S=1024.
// ---------------------------------------------------------------------------

#define L_LAYERS 6
#define EMB      768
#define NHEAD    12
#define HDIM     64
#define FFDIM    3072
#define SEQ      1024
#define NBATCH   2
#define MTOT     (NBATCH * SEQ)   // 2048 rows through every GEMM

typedef __bf16 bf16_t;
typedef __attribute__((ext_vector_type(16))) __bf16 v16bf;
typedef __attribute__((ext_vector_type(8)))  float  v8f;

__device__ __forceinline__ bf16_t f2bf(float f) {
  union { float f; unsigned u; } a; a.f = f;
  unsigned u = a.u + 0x7FFFu + ((a.u >> 16) & 1u);   // RNE
  union { unsigned short s; bf16_t b; } o; o.s = (unsigned short)(u >> 16);
  return o.b;
}

union FragU { v16bf v; unsigned u[8]; bf16_t h[16]; };

__device__ __forceinline__ v8f wmma_bf16(v16bf a, v16bf b, v8f c) {
  return __builtin_amdgcn_wmma_f32_16x16x32_bf16(false, a, false, b,
                                                 (short)0, c, false, false);
}

// Async copy 16B (8 bf16) per lane from global to LDS. Tracked by ASYNCcnt.
// Generic LDS pointer low 32 bits == LDS byte address (ISA 10.2 aperture calc).
__device__ __forceinline__ void async_b128(const void* g, void* l) {
  unsigned lofs = (unsigned)(size_t)l;
  unsigned long long ga = (unsigned long long)(size_t)g;
  asm volatile("global_load_async_to_lds_b128 %0, %1, off"
               :: "v"(lofs), "v"(ga) : "memory");
}
__device__ __forceinline__ void wait_async0() {
  asm volatile("s_wait_asynccnt 0" ::: "memory");
}

// A fragment: 16x32 tile of a row-major matrix (row stride `ld` elements).
// Layout: lanes 0-15 row M=lane, K=0..15; lanes 16-31 K=16..31.
__device__ __forceinline__ v16bf load_frag_a(const bf16_t* base, int ld,
                                             int row0, int k0) {
  int lane = threadIdx.x & 31;
  const bf16_t* p = base + (size_t)(row0 + (lane & 15)) * ld + k0
                         + ((lane >> 4) ? 16 : 0);
  FragU f;
#pragma unroll
  for (int v = 0; v < 8; ++v)
    f.u[v] = *(const unsigned*)(p + 2 * v);
  return f.v;
}

// B fragment from a K-contiguous (N-major) matrix: row n holds K contiguously.
// B layout used here: lane=col N, lanes 0-15 K=0..15 (2/VGPR), lanes 16-31 K=16..31.
__device__ __forceinline__ v16bf load_frag_bt(const bf16_t* base, int ld,
                                              int n0, int k0) {
  int lane = threadIdx.x & 31;
  const bf16_t* p = base + (size_t)(n0 + (lane & 15)) * ld + k0
                         + ((lane >> 4) ? 16 : 0);
  FragU f;
#pragma unroll
  for (int v = 0; v < 8; ++v)
    f.u[v] = *(const unsigned*)(p + 2 * v);
  return f.v;
}

// B fragment from a genuinely row-major KxN matrix (V in attention).
__device__ __forceinline__ v16bf load_frag_b_rm(const bf16_t* base, int ld,
                                                int k0, int n0) {
  int lane = threadIdx.x & 31;
  const bf16_t* p = base + (size_t)(k0 + ((lane >> 4) ? 16 : 0)) * ld
                         + n0 + (lane & 15);
  FragU f;
#pragma unroll
  for (int v = 0; v < 8; ++v) {
    f.h[2 * v]     = p[(size_t)(2 * v) * ld];
    f.h[2 * v + 1] = p[(size_t)(2 * v + 1) * ld];
  }
  return f.v;
}

// ---------------------------------------------------------------------------
// Weight convert fp32 [L][K][N] -> bf16 [L][N][K] via LDS 32x32 tile
// transpose: coalesced reads AND writes, no integer division.
// grid = (K/32, N/32, L), block = 256 (32 cols x 8 rows).
// ---------------------------------------------------------------------------
__global__ void __launch_bounds__(256)
conv_wt(const float* __restrict__ W, bf16_t* __restrict__ Wt, int K, int N) {
  __shared__ float t[32][33];
  const int k0 = blockIdx.x * 32, n0 = blockIdx.y * 32;
  const size_t lw = (size_t)blockIdx.z * K * N;
  const int c = threadIdx.x & 31, r = threadIdx.x >> 5;   // 32 x 8
#pragma unroll
  for (int i = 0; i < 4; ++i)
    t[r + 8 * i][c] = W[lw + (size_t)(k0 + r + 8 * i) * N + n0 + c];
  __syncthreads();
#pragma unroll
  for (int i = 0; i < 4; ++i)
    Wt[lw + (size_t)(n0 + r + 8 * i) * K + k0 + c] = f2bf(t[c][r + 8 * i]);
}

// ---------------------------------------------------------------------------
// LayerNorm over E=768 (block = 256 threads, 3 elems/thread)
// ---------------------------------------------------------------------------
__device__ __forceinline__ float block_sum(float v, float* red) {
  int lane = threadIdx.x & 31, wave = threadIdx.x >> 5;
#pragma unroll
  for (int o = 16; o >= 1; o >>= 1) v += __shfl_xor(v, o, 32);
  if (lane == 0) red[wave] = v;
  __syncthreads();
  float t = (threadIdx.x < 8) ? red[threadIdx.x] : 0.f;
  if (wave == 0) {
#pragma unroll
    for (int o = 4; o >= 1; o >>= 1) t += __shfl_xor(t, o, 32);
    if (lane == 0) red[0] = t;
  }
  __syncthreads();
  float r = red[0];
  __syncthreads();
  return r;
}

template <bool BF16OUT>
__global__ void __launch_bounds__(256)
layernorm_k(const float* __restrict__ x, const float* __restrict__ g,
            const float* __restrict__ b, bf16_t* __restrict__ yb,
            float* __restrict__ yf) {
  __shared__ float red[8];
  int row = blockIdx.x;
  const float* xr = x + (size_t)row * EMB;
  float v[3];
#pragma unroll
  for (int i = 0; i < 3; ++i) v[i] = xr[threadIdx.x + 256 * i];
  float mean = block_sum(v[0] + v[1] + v[2], red) * (1.f / EMB);
  float d2 = 0.f;
#pragma unroll
  for (int i = 0; i < 3; ++i) { float d = v[i] - mean; d2 += d * d; }
  float rstd = rsqrtf(block_sum(d2, red) * (1.f / EMB) + 1e-5f);
#pragma unroll
  for (int i = 0; i < 3; ++i) {
    int c = threadIdx.x + 256 * i;
    float y = (v[i] - mean) * rstd * g[c] + b[c];
    if (BF16OUT) yb[(size_t)row * EMB + c] = f2bf(y);
    else         yf[(size_t)row * EMB + c] = y;
  }
}

// ---------------------------------------------------------------------------
// GEMM: C[M,N] = A[M,K](bf16 row-major) x Bt[N,K](bf16) (+bias,+gelu,+resid)
// Block = 8 waves; block tile 64xBN; wave tile 32x(BN/4).
// K-step 32 double-buffered in LDS via global_load_async_to_lds_b128.
// FLAGS: 1=bias, 2=gelu(exact), 4=residual accumulate fp32, 8=bf16 store
// BN: 256 (wide outputs) or 128 (narrow outputs, 2x more blocks)
// ---------------------------------------------------------------------------
#define APAD 40   // padded LDS row stride (bf16): 80B = 16B-aligned, 20 banks

template <int FLAGS>
__device__ __forceinline__ void store_tile(const v8f& acc, int mbase, int nbase,
                                           const float* bias, float* Cf,
                                           bf16_t* Cb, int N) {
  int lane = threadIdx.x & 31;
  int col = nbase + (lane & 15);
  int rofs = (lane >> 4) * 8;
  float bi = (FLAGS & 1) ? bias[col] : 0.f;
#pragma unroll
  for (int r = 0; r < 8; ++r) {
    int row = mbase + rofs + r;
    float v = acc[r] + bi;
    if (FLAGS & 2) v = 0.5f * v * (1.f + erff(v * 0.70710678118654752f));
    if (FLAGS & 4)      Cf[(size_t)row * N + col] += v;
    else if (FLAGS & 8) Cb[(size_t)row * N + col] = f2bf(v);
    else                Cf[(size_t)row * N + col] = v;
  }
}

template <int FLAGS, int BN>
__global__ void __launch_bounds__(256)
gemm_bf16_wmma(const bf16_t* __restrict__ A, const bf16_t* __restrict__ Bt,
               const float* __restrict__ bias, float* __restrict__ Cf,
               bf16_t* __restrict__ Cb, int M, int N, int K) {
  constexpr int JT = BN / 64;   // n-frags per wave == B-row chunks per thread
  __shared__ __align__(16) bf16_t As[2][64][APAD];
  __shared__ __align__(16) bf16_t Bs[2][BN][APAD];
  (void)M;
  const int tid = threadIdx.x;
  const int wave = tid >> 5;
  const int mblk = blockIdx.y * 64;
  const int nblk = blockIdx.x * BN;
  const int mw = (wave >> 2) * 32;
  const int nw = (wave & 3) * (BN / 4);

  const int arow = tid >> 2;          // 0..63
  const int acol = (tid & 3) * 8;     // 0,8,16,24

  auto fill = [&](int buf, int kk) {
    async_b128(A + (size_t)(mblk + arow) * K + kk + acol, &As[buf][arow][acol]);
#pragma unroll
    for (int t = 0; t < JT; ++t)
      async_b128(Bt + (size_t)(nblk + arow + 64 * t) * K + kk + acol,
                 &Bs[buf][arow + 64 * t][acol]);
  };

  v8f acc[2][JT] = {};
  fill(0, 0);
  wait_async0();
  __syncthreads();

  const int nk = K / 32;
  for (int it = 0; it < nk; ++it) {
    if (it + 1 < nk) fill((it + 1) & 1, (it + 1) * 32);
    const bf16_t* Al = &As[it & 1][0][0];
    const bf16_t* Bl = &Bs[it & 1][0][0];
    v16bf fa0 = load_frag_a(Al, APAD, mw, 0);
    v16bf fa1 = load_frag_a(Al, APAD, mw + 16, 0);
#pragma unroll
    for (int j = 0; j < JT; ++j) {
      v16bf fb = load_frag_bt(Bl, APAD, nw + 16 * j, 0);
      acc[0][j] = wmma_bf16(fa0, fb, acc[0][j]);
      acc[1][j] = wmma_bf16(fa1, fb, acc[1][j]);
    }
    wait_async0();
    __syncthreads();
  }
#pragma unroll
  for (int i = 0; i < 2; ++i)
#pragma unroll
    for (int j = 0; j < JT; ++j)
      store_tile<FLAGS>(acc[i][j], mblk + mw + 16 * i, nblk + nw + 16 * j,
                        bias, Cf, Cb, N);
}

// ---------------------------------------------------------------------------
// Flash attention: block = 8 waves sharing one (b,h); wave owns a 16-query
// tile. K/V 32x64 tiles async-double-buffered in LDS; QK^T, P.V and softmax
// row sums (P x ones) via bf16 WMMA; row max via shfl butterfly.
// ---------------------------------------------------------------------------
#define KVPAD 72   // 144B rows: 16B aligned, 36 banks -> conflict-free
#define LDSP  40

__global__ void __launch_bounds__(256)
attention_k(const bf16_t* __restrict__ qkv, const float* __restrict__ amask,
            bf16_t* __restrict__ out) {
  __shared__ __align__(16) bf16_t Ks[2][32][KVPAD];
  __shared__ __align__(16) bf16_t Vs[2][32][KVPAD];
  __shared__ __align__(16) bf16_t Pls[8][16 * LDSP];

  const int tid = threadIdx.x;
  const int wave = tid >> 5, lane = tid & 31;
  const int bh = blockIdx.x >> 3;            // 0..23
  const int qc = blockIdx.x & 7;             // 128-query chunk
  const int b = bh / NHEAD, h = bh % NHEAD;
  const int q0 = qc * 128 + wave * 16;
  const size_t t0 = (size_t)b * SEQ;
  const int R3 = 3 * EMB;
  const bf16_t* Qg = qkv + t0 * R3 + h * HDIM;
  const bf16_t* Kg = Qg + EMB;
  const bf16_t* Vg = Qg + 2 * EMB;
  const float* maskb = amask + (size_t)b * SEQ;

  // 32x64 tile = 256 16B-chunks: one K chunk + one V chunk per thread
  const int crow = tid >> 3;        // 0..31 (kv row)
  const int ccol = (tid & 7) * 8;   // 0..56 (d)
  auto fill_kv = [&](int buf, int kv0) {
    async_b128(Kg + (size_t)(kv0 + crow) * R3 + ccol, &Ks[buf][crow][ccol]);
    async_b128(Vg + (size_t)(kv0 + crow) * R3 + ccol, &Vs[buf][crow][ccol]);
  };

  const v16bf aQ0 = load_frag_a(Qg, R3, q0, 0);
  const v16bf aQ1 = load_frag_a(Qg, R3, q0, 32);

  FragU fo;
#pragma unroll
  for (int i = 0; i < 16; ++i) fo.h[i] = f2bf(1.f);
  const v16bf onesB = fo.v;

  float mrow[8], lrow[8];
  v8f O0 = {}, O1 = {}, O2 = {}, O3 = {};
#pragma unroll
  for (int r = 0; r < 8; ++r) { mrow[r] = -1e30f; lrow[r] = 0.f; }

  bf16_t* lp = Pls[wave];
  const int rofs = (lane >> 4) * 8;

  fill_kv(0, 0);
  wait_async0();
  __syncthreads();

  for (int it = 0; it < SEQ / 32; ++it) {
    const int kv = it * 32;
    const int buf = it & 1;
    if (it + 1 < SEQ / 32) fill_kv(buf ^ 1, kv + 32);

    const bf16_t* Kl = &Ks[buf][0][0];
    const bf16_t* Vl = &Vs[buf][0][0];

    // scores: two 16x16 tiles of (Q K^T)/8 + mask
    v8f s0 = {}, s1 = {};
    s0 = wmma_bf16(aQ0, load_frag_bt(Kl, KVPAD, 0, 0),   s0);
    s0 = wmma_bf16(aQ1, load_frag_bt(Kl, KVPAD, 0, 32),  s0);
    s1 = wmma_bf16(aQ0, load_frag_bt(Kl, KVPAD, 16, 0),  s1);
    s1 = wmma_bf16(aQ1, load_frag_bt(Kl, KVPAD, 16, 32), s1);
    const float madd0 = (1.f - maskb[kv + (lane & 15)])      * -10000.f;
    const float madd1 = (1.f - maskb[kv + 16 + (lane & 15)]) * -10000.f;

    float alpha[8];
#pragma unroll
    for (int r = 0; r < 8; ++r) {
      float x0 = s0[r] * 0.125f + madd0;
      float x1 = s1[r] * 0.125f + madd1;
      float mx = fmaxf(x0, x1);
#pragma unroll
      for (int o = 8; o >= 1; o >>= 1) mx = fmaxf(mx, __shfl_xor(mx, o, 32));
      float mnew = fmaxf(mrow[r], mx);
      alpha[r] = __expf(mrow[r] - mnew);
      mrow[r] = mnew;
      // exponentials -> LDS (P tile, A-fragment staging)
      lp[(rofs + r) * LDSP + (lane & 15)]      = f2bf(__expf(x0 - mnew));
      lp[(rofs + r) * LDSP + 16 + (lane & 15)] = f2bf(__expf(x1 - mnew));
    }
#pragma unroll
    for (int r = 0; r < 8; ++r) {
      O0[r] *= alpha[r]; O1[r] *= alpha[r];
      O2[r] *= alpha[r]; O3[r] *= alpha[r];
    }
    asm volatile("s_wait_dscnt 0" ::: "memory");
    const v16bf aP = load_frag_a(lp, LDSP, 0, 0);

    // row sums via WMMA: l_part = P x ones (replicated across columns)
    v8f ls = {};
    ls = wmma_bf16(aP, onesB, ls);
    // O += P x V
    O0 = wmma_bf16(aP, load_frag_b_rm(Vl, KVPAD, 0, 0),  O0);
    O1 = wmma_bf16(aP, load_frag_b_rm(Vl, KVPAD, 0, 16), O1);
    O2 = wmma_bf16(aP, load_frag_b_rm(Vl, KVPAD, 0, 32), O2);
    O3 = wmma_bf16(aP, load_frag_b_rm(Vl, KVPAD, 0, 48), O3);
#pragma unroll
    for (int r = 0; r < 8; ++r) lrow[r] = lrow[r] * alpha[r] + ls[r];

    wait_async0();
    __syncthreads();
  }

#pragma unroll
  for (int r = 0; r < 8; ++r) {
    float inv = 1.f / lrow[r];
    int srow = q0 + rofs + r;
    bf16_t* orow = out + (t0 + srow) * (size_t)EMB + h * HDIM + (lane & 15);
    orow[0]  = f2bf(O0[r] * inv);
    orow[16] = f2bf(O1[r] * inv);
    orow[32] = f2bf(O2[r] * inv);
    orow[48] = f2bf(O3[r] * inv);
  }
}

// ---------------------------------------------------------------------------
// Host orchestration
// ---------------------------------------------------------------------------
extern "C" void kernel_launch(void* const* d_in, const int* in_sizes, int n_in,
                              void* d_out, int out_size, void* d_ws,
                              size_t ws_size, hipStream_t stream) {
  (void)in_sizes; (void)n_in; (void)out_size; (void)ws_size;
  const float* emb   = (const float*)d_in[0];
  const float* amask = (const float*)d_in[1];
  const float* Wqkv  = (const float*)d_in[2];
  const float* bqkv  = (const float*)d_in[3];
  const float* Wo    = (const float*)d_in[4];
  const float* bo    = (const float*)d_in[5];
  const float* ln1g  = (const float*)d_in[6];
  const float* ln1b  = (const float*)d_in[7];
  const float* ln2g  = (const float*)d_in[8];
  const float* ln2b  = (const float*)d_in[9];
  const float* Wfc   = (const float*)d_in[10];
  const float* bfc   = (const float*)d_in[11];
  const float* Wmo   = (const float*)d_in[12];
  const float* bmo   = (const float*)d_in[13];
  const float* lnfg  = (const float*)d_in[14];
  const float* lnfb  = (const float*)d_in[15];

  char* ws = (char*)d_ws;
  size_t off = 0;
  auto take = [&](size_t bytes) -> char* {
    char* p = ws + off;
    off += (bytes + 255) & ~(size_t)255;
    return p;
  };
  bf16_t* WqkvT = (bf16_t*)take((size_t)L_LAYERS * 3 * EMB * EMB * 2);
  bf16_t* WoT   = (bf16_t*)take((size_t)L_LAYERS * EMB * EMB * 2);
  bf16_t* WfcT  = (bf16_t*)take((size_t)L_LAYERS * FFDIM * EMB * 2);
  bf16_t* WmoT  = (bf16_t*)take((size_t)L_LAYERS * EMB * FFDIM * 2);
  float*  x     = (float*)take((size_t)MTOT * EMB * 4);
  bf16_t* hbuf  = (bf16_t*)take((size_t)MTOT * EMB * 2);
  bf16_t* qkvb  = (bf16_t*)take((size_t)MTOT * 3 * EMB * 2);
  bf16_t* attnb = (bf16_t*)take((size_t)MTOT * EMB * 2);
  bf16_t* ffb   = (bf16_t*)take((size_t)MTOT * FFDIM * 2);

  conv_wt<<<dim3(EMB / 32, (3 * EMB) / 32, L_LAYERS), 256, 0, stream>>>(
      Wqkv, WqkvT, EMB, 3 * EMB);
  conv_wt<<<dim3(EMB / 32, EMB / 32, L_LAYERS), 256, 0, stream>>>(
      Wo, WoT, EMB, EMB);
  conv_wt<<<dim3(EMB / 32, FFDIM / 32, L_LAYERS), 256, 0, stream>>>(
      Wfc, WfcT, EMB, FFDIM);
  conv_wt<<<dim3(FFDIM / 32, EMB / 32, L_LAYERS), 256, 0, stream>>>(
      Wmo, WmoT, FFDIM, EMB);
  hipMemcpyAsync(x, emb, (size_t)MTOT * EMB * 4, hipMemcpyDeviceToDevice,
                 stream);

  for (int l = 0; l < L_LAYERS; ++l) {
    layernorm_k<true><<<MTOT, 256, 0, stream>>>(x, ln1g + l * EMB,
                                                ln1b + l * EMB, hbuf, nullptr);
    gemm_bf16_wmma<9, 256>
        <<<dim3((3 * EMB) / 256, MTOT / 64), 256, 0, stream>>>(
            hbuf, WqkvT + (size_t)l * 3 * EMB * EMB, bqkv + (size_t)l * 3 * EMB,
            nullptr, qkvb, MTOT, 3 * EMB, EMB);
    attention_k<<<NBATCH * NHEAD * 8, 256, 0, stream>>>(qkvb, amask, attnb);
    gemm_bf16_wmma<5, 128><<<dim3(EMB / 128, MTOT / 64), 256, 0, stream>>>(
        attnb, WoT + (size_t)l * EMB * EMB, bo + (size_t)l * EMB, x, nullptr,
        MTOT, EMB, EMB);
    layernorm_k<true><<<MTOT, 256, 0, stream>>>(x, ln2g + l * EMB,
                                                ln2b + l * EMB, hbuf, nullptr);
    gemm_bf16_wmma<11, 256><<<dim3(FFDIM / 256, MTOT / 64), 256, 0, stream>>>(
        hbuf, WfcT + (size_t)l * FFDIM * EMB, bfc + (size_t)l * FFDIM, nullptr,
        ffb, MTOT, FFDIM, EMB);
    gemm_bf16_wmma<5, 128><<<dim3(EMB / 128, MTOT / 64), 256, 0, stream>>>(
        ffb, WmoT + (size_t)l * EMB * FFDIM, bmo + (size_t)l * EMB, x, nullptr,
        MTOT, EMB, FFDIM);
  }
  layernorm_k<false><<<MTOT, 256, 0, stream>>>(x, lnfg, lnfb, nullptr,
                                               (float*)d_out);
}
